// FoldingNet_85169201480476
// MI455X (gfx1250) — matrix operand order repaired and checked by reading, not compiled, not run on hardware
//
#include <hip/hip_runtime.h>

// ---------------------------------------------------------------------------
// Types / helpers
// ---------------------------------------------------------------------------
typedef __attribute__((ext_vector_type(16))) __bf16 v16bf;
typedef __attribute__((ext_vector_type(8)))  float  v8f;
typedef __attribute__((ext_vector_type(4)))  unsigned int u32x4;
typedef __attribute__((ext_vector_type(8)))  int       i32x8;
typedef __attribute__((ext_vector_type(4)))  int       i32x4;

#define DEVINL __device__ __forceinline__

DEVINL unsigned short f2bf(float f) {
  unsigned u = __float_as_uint(f);
  u = u + 0x7FFFu + ((u >> 16) & 1u);          // round-to-nearest-even
  return (unsigned short)(u >> 16);
}
DEVINL float bf2f(unsigned short h) { return __uint_as_float(((unsigned)h) << 16); }

union FragU { v16bf v; unsigned u[8]; };

// A operand: 16x32 bf16 (ISA 7.12.2): lane%16 = M; VGPR p<4 -> K=2p(+8 hi lanes),
// p>=4 -> K=16+2(p-4)(+8 hi lanes)
DEVINL v16bf load_a_frag(const unsigned short* base, int ld, int kbase, int lane) {
  FragU f;
  const unsigned short* p = base + (size_t)(lane & 15) * ld + kbase;
  const int hoff = ((lane >> 4) & 1) * 8;
  #pragma unroll
  for (int q = 0; q < 8; ++q) {
    int kk = ((q < 4) ? 0 : 16) + hoff + 2 * (q & 3);
    f.u[q] = *(const unsigned*)(p + kk);
  }
  return f.v;
}
// B operand: 32x16 bf16: lane%16 = column (point), K same order as A (hi lanes K+=8)
DEVINL v16bf load_b_frag(const unsigned short* base, int ld, int kbase, int lane) {
  FragU f;
  const unsigned short* p = base + (size_t)(lane & 15) * ld + kbase + ((lane >> 4) & 1) * 8;
  #pragma unroll
  for (int q = 0; q < 8; ++q) f.u[q] = *(const unsigned*)(p + ((q < 4) ? 0 : 16) + 2 * (q & 3));
  return f.v;
}

DEVINL void atomicMaxF(float* addr, float val) {
  unsigned* ua = (unsigned*)addr;
  unsigned old = *ua;
  while (__uint_as_float(old) < val) {
    unsigned assumed = old;
    old = atomicCAS(ua, assumed, __float_as_uint(val));
    if (old == assumed) break;
  }
}

// branchless sorted insertion (ascending d2) into 16-entry register list
DEVINL void topk_insert(float (&bd)[16], int (&bi)[16], float d, int ci) {
  float cd = d; int cix = ci;
  #pragma unroll
  for (int j = 0; j < 16; ++j) {
    bool sw = cd < bd[j];
    float td = bd[j]; int ti = bi[j];
    bd[j] = sw ? cd : td;  bi[j] = sw ? cix : ti;
    cd   = sw ? td : cd;   cix  = sw ? ti : cix;
  }
}

// ---------------------------------------------------------------------------
// Tensor Data Mover: 2-D bf16 tile (tileRows x dim0 elements) global -> LDS.
// D# per CDNA5 ISA 8.3/8.4: group0 = {count/type/lds/global addr},
// group1 = {data_size, dims, tile dims, strides}. Dims == tile (fully in-bounds).
// 6-arg builtin form on this toolchain: (g0, g1, g2, g3, g4, cpol).
// ---------------------------------------------------------------------------
DEVINL void tdm_load_2d_bf16(unsigned ldsAddr, const void* gsrc,
                             unsigned dim0, unsigned tileRows) {
  unsigned long long ga = (unsigned long long)(size_t)gsrc;
  u32x4 g0;
  g0[0] = 1u;                                        // count=1 (valid user desc)
  g0[1] = ldsAddr;                                   // lds_addr (bytes)
  g0[2] = (unsigned)(ga & 0xffffffffu);              // global_addr[31:0]
  g0[3] = (unsigned)((ga >> 32) & 0x01ffffffu)       // global_addr[56:32]
        | (2u << 30);                                // type = 2 ("image")
  i32x8 g1;
  g1[0] = (int)(1u << 16);                           // data_size=1 (2 bytes)
  g1[1] = (int)((dim0 & 0xffffu) << 16);             // tensor_dim0[15:0]
  g1[2] = (int)(((dim0 >> 16) & 0xffffu)             // tensor_dim0[31:16]
        | ((tileRows & 0xffffu) << 16));             // tensor_dim1[15:0]
  g1[3] = (int)(((tileRows >> 16) & 0xffffu)         // tensor_dim1[31:16]
        | ((dim0 & 0xffffu) << 16));                 // tile_dim0
  g1[4] = (int)(tileRows & 0xffffu);                 // tile_dim1 (tile_dim2=0)
  g1[5] = (int)dim0;                                 // tensor_dim0_stride[31:0]
  g1[6] = 0;                                         // stride0[47:32] | stride1[15:0]
  g1[7] = 0;
  i32x4 z;
  z[0] = 0; z[1] = 0; z[2] = 0; z[3] = 0;
  i32x8 z8;
  #pragma unroll
  for (int i = 0; i < 8; ++i) z8[i] = 0;
  __builtin_amdgcn_tensor_load_to_lds(g0, g1, z, z, z8, 0);
}

// ---------------------------------------------------------------------------
// Weight pack: f32 [Cout][Cin] -> bf16 [CoutPad][KPAD], zero padded
// ---------------------------------------------------------------------------
__global__ void pack_weight_k(const float* __restrict__ W, unsigned short* __restrict__ Wp,
                              int Cout, int Cin, int CoutPad, int KPAD) {
  int i = blockIdx.x * 256 + threadIdx.x;
  if (i >= CoutPad * KPAD) return;
  int r = i / KPAD, k = i % KPAD;
  float v = (r < Cout && k < Cin) ? W[(size_t)r * Cin + k] : 0.0f;
  Wp[i] = f2bf(v);
}

// ---------------------------------------------------------------------------
// Wide 1x1-conv GEMM: wave computes 16(cout) x 64(points); A frag reused 4x.
// MODE 0: bf16 store (Cout==CoutPad). MODE 1: f32 store, guard cout<Cout.
// MODE 2: fused global-max epilogue (atomic), Cout==CoutPad.
// ---------------------------------------------------------------------------
template<int KPAD, int MODE>
__global__ __launch_bounds__(32)
void conv_wide_wmma(const unsigned short* __restrict__ X, const unsigned short* __restrict__ W,
                    const float* __restrict__ bias,
                    unsigned short* __restrict__ Y, int ldY,
                    float* __restrict__ Yf, int ldYf,
                    float* __restrict__ maxOut, int maxN,
                    int Cout, int doRelu) {
  const int lane = threadIdx.x & 31;
  const int nBase = blockIdx.x * 64;
  const int coutBase = blockIdx.y * 16;

  v8f acc[4];
  #pragma unroll
  for (int t = 0; t < 4; ++t)
    #pragma unroll
    for (int i = 0; i < 8; ++i) acc[t][i] = 0.0f;

  const unsigned short* Wt = W + (size_t)coutBase * KPAD;
  const unsigned short* Xt = X + (size_t)nBase * KPAD;
  #pragma unroll
  for (int kb = 0; kb < KPAD; kb += 32) {
    v16bf a = load_a_frag(Wt, KPAD, kb, lane);
    #pragma unroll
    for (int t = 0; t < 4; ++t) {
      v16bf b = load_b_frag(Xt + (size_t)(t * 16) * KPAD, KPAD, kb, lane);
      acc[t] = __builtin_amdgcn_wmma_f32_16x16x32_bf16(false, a, false, b,
                                                       (short)0, acc[t], false, false);
    }
  }

  const int hi = (lane >> 4) & 1;
  #pragma unroll
  for (int t = 0; t < 4; ++t) {
    const int n = nBase + t * 16 + (lane & 15);
    #pragma unroll
    for (int v = 0; v < 8; ++v) {
      int cout = coutBase + v + hi * 8;
      if (MODE == 0) {
        float val = acc[t][v] + bias[cout];
        if (doRelu) val = fmaxf(val, 0.0f);
        Y[(size_t)n * ldY + cout] = f2bf(val);
      } else if (MODE == 1) {
        if (cout < Cout) {
          float val = acc[t][v] + bias[cout];
          if (doRelu) val = fmaxf(val, 0.0f);
          Yf[(size_t)n * ldYf + cout] = val;
        }
      } else {
        float val = acc[t][v] + bias[cout];
        atomicMaxF(&maxOut[(size_t)(n / maxN) * 1024 + cout], val);
      }
    }
  }
}

template<int KPAD, int MODE>
static void conv_wide(const unsigned short* X, const unsigned short* W, const float* bias,
                      unsigned short* Y, int ldY, float* Yf, int ldYf,
                      float* maxOut, int maxN,
                      int nRows, int Cout, int CoutPad, int relu, hipStream_t s) {
  dim3 g(nRows / 64, CoutPad / 16);
  conv_wide_wmma<KPAD, MODE><<<g, 32, 0, s>>>(X, W, bias, Y, ldY, Yf, ldYf, maxOut, maxN,
                                              Cout, relu);
}

// Narrow variant for tiny row counts (16 rows): 16x16 tile per wave, bf16 out.
template<int KPAD>
__global__ __launch_bounds__(32)
void conv_small_wmma(const unsigned short* __restrict__ X, const unsigned short* __restrict__ W,
                     const float* __restrict__ bias, unsigned short* __restrict__ Y,
                     int ldY, int doRelu) {
  const int lane = threadIdx.x & 31;
  const int nBase = blockIdx.x * 16;
  const int coutBase = blockIdx.y * 16;
  v8f acc;
  #pragma unroll
  for (int i = 0; i < 8; ++i) acc[i] = 0.0f;
  const unsigned short* Wt = W + (size_t)coutBase * KPAD;
  const unsigned short* Xt = X + (size_t)nBase * KPAD;
  #pragma unroll
  for (int kb = 0; kb < KPAD; kb += 32) {
    v16bf a = load_a_frag(Wt, KPAD, kb, lane);
    v16bf b = load_b_frag(Xt, KPAD, kb, lane);
    acc = __builtin_amdgcn_wmma_f32_16x16x32_bf16(false, a, false, b, (short)0, acc, false, false);
  }
  const int hi = (lane >> 4) & 1;
  const int n = nBase + (lane & 15);
  #pragma unroll
  for (int v = 0; v < 8; ++v) {
    int cout = coutBase + v + hi * 8;
    float val = acc[v] + bias[cout];
    if (doRelu) val = fmaxf(val, 0.0f);
    Y[(size_t)n * ldY + cout] = f2bf(val);
  }
}

// ---------------------------------------------------------------------------
// kNN on raw xyz points: thread-per-query, LDS candidate tiles
// ---------------------------------------------------------------------------
__global__ __launch_bounds__(256)
void knn_xyz_k(const float* __restrict__ pts, int* __restrict__ idxOut, int N) {
  __shared__ float sx[256], sy[256], sz[256];
  int tid = threadIdx.x;
  int q = blockIdx.x * 256 + tid;
  int b = q / N, n = q % N;
  const float* pb = pts + (size_t)b * N * 3;
  float px = pb[n * 3 + 0], py = pb[n * 3 + 1], pz = pb[n * 3 + 2];

  float bd[16]; int bi[16];
  #pragma unroll
  for (int j = 0; j < 16; ++j) { bd[j] = 3.402823466e38f; bi[j] = 0; }

  for (int c0 = 0; c0 < N; c0 += 256) {
    __syncthreads();
    sx[tid] = pb[(c0 + tid) * 3 + 0];
    sy[tid] = pb[(c0 + tid) * 3 + 1];
    sz[tid] = pb[(c0 + tid) * 3 + 2];
    __syncthreads();
    for (int j = 0; j < 256; ++j) {
      float dx = px - sx[j], dy = py - sy[j], dz = pz - sz[j];
      float d2 = dx * dx + dy * dy + dz * dz;
      if (d2 < bd[15]) topk_insert(bd, bi, d2, c0 + j);
    }
  }
  #pragma unroll
  for (int j = 0; j < 16; ++j) idxOut[(size_t)q * 16 + j] = bi[j];
}

// ---------------------------------------------------------------------------
// kNN on bf16 features via WMMA score tiles. 8 waves/WG, 16 queries/wave.
// Candidate tiles (64 rows x C) DMA'd into double-buffered LDS by the TDM;
// wave 0 issues tensor_load_to_lds and synchronizes via TENSORcnt + barrier.
// ---------------------------------------------------------------------------
template<int C>
__global__ __launch_bounds__(256)
void knn_feat_wmma(const unsigned short* __restrict__ X, const float* __restrict__ sq,
                   int* __restrict__ idxOut, int N) {
  __shared__ unsigned short s_cand[2][64 * C];
  __shared__ float s_sc[8][16][17];
  __shared__ float s_csq[64];

  const int tid = threadIdx.x, lane = tid & 31, wave = tid >> 5;
  const int tilesPerBatch = N / 128;
  const int b = blockIdx.x / tilesPerBatch;
  const int qt = blockIdx.x % tilesPerBatch;
  const size_t rowBase = (size_t)b * N;
  const int qbase = qt * 128 + wave * 16;
  const unsigned short* Xb = X + rowBase * C;

  v16bf afr[C / 32];
  #pragma unroll
  for (int kb = 0; kb < C; kb += 32)
    afr[kb / 32] = load_a_frag(Xb + (size_t)qbase * C, C, kb, lane);
  const float sqq = sq[rowBase + qbase + (lane & 15)];

  float bd[16]; int bi[16];
  #pragma unroll
  for (int j = 0; j < 16; ++j) { bd[j] = 3.402823466e38f; bi[j] = 0; }

  // prologue: DMA first candidate tile
  if (wave == 0)
    tdm_load_2d_bf16((unsigned)(size_t)&s_cand[0][0], Xb, C, 64);

  for (int ct = 0; ct < N; ct += 64) {
    const int cur = (ct >> 6) & 1;
    if (wave == 0) __builtin_amdgcn_s_wait_tensorcnt(0);
    __syncthreads();                                  // cur tile visible; prev reads done
    if (wave == 0 && ct + 64 < N)
      tdm_load_2d_bf16((unsigned)(size_t)&s_cand[cur ^ 1][0],
                       Xb + (size_t)(ct + 64) * C, C, 64);
    if (tid < 64) s_csq[tid] = sq[rowBase + ct + tid];
    if (ct + 64 < N)
      __builtin_prefetch((const void*)(Xb + (size_t)(ct + 64) * C + tid * 16), 0, 0);

    const unsigned short* cb = &s_cand[cur][0];
    for (int st = 0; st < 4; ++st) {
      v8f acc;
      #pragma unroll
      for (int i = 0; i < 8; ++i) acc[i] = 0.0f;
      #pragma unroll
      for (int kb = 0; kb < C; kb += 32) {
        v16bf bfr = load_b_frag(cb + (size_t)(st * 16) * C, C, kb, lane);
        acc = __builtin_amdgcn_wmma_f32_16x16x32_bf16(false, afr[kb / 32], false, bfr,
                                                      (short)0, acc, false, false);
      }
      const int hi = (lane >> 4) & 1;
      #pragma unroll
      for (int v = 0; v < 8; ++v) s_sc[wave][v + hi * 8][lane & 15] = acc[v];
      __syncthreads();
      if (lane < 16) {
        for (int j = 0; j < 16; ++j) {
          float inner = s_sc[wave][lane][j];
          float d2 = sqq + s_csq[st * 16 + j] - 2.0f * inner;
          int ci = ct + st * 16 + j;
          if (d2 < bd[15]) topk_insert(bd, bi, d2, ci);
        }
      }
      __syncthreads();
    }
  }
  if (lane < 16) {
    size_t row = rowBase + qbase + lane;
    #pragma unroll
    for (int j = 0; j < 16; ++j) idxOut[row * 16 + j] = bi[j];
  }
}

// ---------------------------------------------------------------------------
// Misc glue kernels
// ---------------------------------------------------------------------------
__global__ void sqnorm_k(const unsigned short* __restrict__ X, float* __restrict__ sq,
                         int C, int total) {
  int i = blockIdx.x * 256 + threadIdx.x;
  if (i >= total) return;
  const unsigned short* r = X + (size_t)i * C;
  float s = 0.0f;
  for (int c = 0; c < C; ++c) { float v = bf2f(r[c]); s += v * v; }
  sq[i] = s;
}

__global__ void local_cov_k(const float* __restrict__ pts, const int* __restrict__ idx,
                            unsigned short* __restrict__ X0, int N) {
  int q = blockIdx.x * 256 + threadIdx.x;
  int b = q / N, n = q % N;
  const float* pb = pts + (size_t)b * N * 3;
  int i0 = idx[(size_t)q * 16 + 0], i1 = idx[(size_t)q * 16 + 1];
  float p0[3] = { pb[n * 3], pb[n * 3 + 1], pb[n * 3 + 2] };
  float a0[3] = { pb[i0 * 3], pb[i0 * 3 + 1], pb[i0 * 3 + 2] };
  float a1[3] = { pb[i1 * 3], pb[i1 * 3 + 1], pb[i1 * 3 + 2] };
  unsigned short* row = X0 + (size_t)q * 32;
  row[0] = f2bf(p0[0]); row[1] = f2bf(p0[1]); row[2] = f2bf(p0[2]);
  #pragma unroll
  for (int r = 0; r < 3; ++r)
    #pragma unroll
    for (int c = 0; c < 3; ++c) row[3 + r * 3 + c] = f2bf(a0[r] * a1[c]);
  for (int k = 12; k < 32; ++k) row[k] = 0;
}

__global__ void local_maxpool_k(const unsigned short* __restrict__ X, const int* __restrict__ idx,
                                unsigned short* __restrict__ Y, int C, int N) {
  int gid = blockIdx.x * 256 + threadIdx.x;
  int n = gid / C, c = gid % C;
  int b = n / N;
  float m = -3.402823466e38f;
  const int* id = idx + (size_t)n * 16;
  #pragma unroll
  for (int j = 0; j < 16; ++j) {
    int nb = id[j];
    m = fmaxf(m, bf2f(X[((size_t)b * N + nb) * C + c]));
  }
  Y[(size_t)n * C + c] = f2bf(m);
}

__global__ void fill_f32_k(float* p, int n, float v) {
  int i = blockIdx.x * 256 + threadIdx.x;
  if (i < n) p[i] = v;
}

__global__ void glob_to_bf_k(const float* __restrict__ gf, unsigned short* __restrict__ gb) {
  int i = blockIdx.x * 256 + threadIdx.x;           // 16*1024
  int r = i / 1024;
  gb[i] = (r < 4) ? f2bf(gf[i]) : (unsigned short)0;
}

__global__ void build_fold_input_k(const unsigned short* __restrict__ code,
                                   const float* __restrict__ mid,
                                   unsigned short* __restrict__ dc, int useGrid) {
  int row = blockIdx.x * 256 + threadIdx.x;
  if (row >= 4 * 2032) return;
  int b = row / 2032, m = row % 2032;
  int mm = (m < 2025) ? m : 2024;
  unsigned short* d = dc + (size_t)row * 544;
  const unsigned short* c = code + (size_t)b * 512;
  for (int k = 0; k < 512; ++k) d[k] = c[k];
  if (useGrid) {
    float step = 0.6f / 44.0f;
    float gx = -0.3f + step * (float)(mm / 45);
    float gy = -0.3f + step * (float)(mm % 45);
    d[512] = f2bf(gx); d[513] = f2bf(gy);
    for (int k = 514; k < 544; ++k) d[k] = 0;
  } else {
    const float* mr = mid + (size_t)row * 4;
    d[512] = f2bf(mr[0]); d[513] = f2bf(mr[1]); d[514] = f2bf(mr[2]);
    for (int k = 515; k < 544; ++k) d[k] = 0;
  }
}

__global__ void writeout_k(const float* __restrict__ rec, float* __restrict__ out) {
  int i = blockIdx.x * 256 + threadIdx.x;
  if (i >= 4 * 2025) return;
  int b = i / 2025, m = i % 2025;
  const float* r = rec + ((size_t)b * 2032 + m) * 4;
  out[(size_t)i * 3 + 0] = r[0];
  out[(size_t)i * 3 + 1] = r[1];
  out[(size_t)i * 3 + 2] = r[2];
}

// ---------------------------------------------------------------------------
// Host launch
// ---------------------------------------------------------------------------
extern "C" void kernel_launch(void* const* d_in, const int* in_sizes, int n_in,
                              void* d_out, int out_size, void* d_ws, size_t ws_size,
                              hipStream_t stream) {
  (void)in_sizes; (void)n_in; (void)out_size; (void)ws_size;
  const int N = 8192, BN = 4 * 8192;
  const int DR = 4 * 2032;

  const float* pts = (const float*)d_in[0];
  const float* b1  = (const float*)d_in[2];
  const float* b2  = (const float*)d_in[4];
  const float* b3  = (const float*)d_in[6];
  const float* g1ab= (const float*)d_in[8];
  const float* g1bb= (const float*)d_in[10];
  const float* g2ab= (const float*)d_in[12];
  const float* g2bb= (const float*)d_in[14];
  const float* m1b = (const float*)d_in[16];
  const float* m2b = (const float*)d_in[18];
  const float* f1ab= (const float*)d_in[20];
  const float* f1bb= (const float*)d_in[22];
  const float* f1cb= (const float*)d_in[24];
  const float* f2ab= (const float*)d_in[26];
  const float* f2bb= (const float*)d_in[28];
  const float* f2cb= (const float*)d_in[30];

  char* w = (char*)d_ws; size_t off = 0;
  auto alloc = [&](size_t bytes) -> void* {
    void* p = w + off; off = (off + bytes + 255) & ~(size_t)255; return p;
  };
  typedef unsigned short us;
  us* w1p  = (us*)alloc((size_t)64 * 32 * 2);
  us* w2p  = (us*)alloc((size_t)64 * 64 * 2);
  us* w3p  = (us*)alloc((size_t)64 * 64 * 2);
  us* g1ap = (us*)alloc((size_t)64 * 64 * 2);
  us* g1bp = (us*)alloc((size_t)128 * 64 * 2);
  us* g2ap = (us*)alloc((size_t)128 * 128 * 2);
  us* g2bp = (us*)alloc((size_t)1024 * 128 * 2);
  us* m1p  = (us*)alloc((size_t)512 * 1024 * 2);
  us* m2p  = (us*)alloc((size_t)512 * 512 * 2);
  us* f1ap = (us*)alloc((size_t)512 * 544 * 2);
  us* f1bp = (us*)alloc((size_t)512 * 512 * 2);
  us* f1cp = (us*)alloc((size_t)16 * 512 * 2);
  us* f2ap = (us*)alloc((size_t)512 * 544 * 2);
  us* f2bp = (us*)alloc((size_t)512 * 512 * 2);
  us* f2cp = (us*)alloc((size_t)16 * 512 * 2);

  int* idx1 = (int*)alloc((size_t)BN * 16 * 4);
  int* idx2 = (int*)alloc((size_t)BN * 16 * 4);
  int* idx3 = (int*)alloc((size_t)BN * 16 * 4);
  us* X0 = (us*)alloc((size_t)BN * 32 * 2);
  us* X1 = (us*)alloc((size_t)BN * 64 * 2);
  us* X2 = (us*)alloc((size_t)BN * 64 * 2);
  us* X3 = (us*)alloc((size_t)BN * 64 * 2);
  float* sqx3 = (float*)alloc((size_t)BN * 4);
  us* P1 = (us*)alloc((size_t)BN * 64 * 2);
  us* G1 = (us*)alloc((size_t)BN * 64 * 2);
  us* G  = (us*)alloc((size_t)BN * 128 * 2);
  float* sqg = (float*)alloc((size_t)BN * 4);
  us* P2 = (us*)alloc((size_t)BN * 128 * 2);
  us* H1 = (us*)alloc((size_t)BN * 128 * 2);
  float* globf = (float*)alloc((size_t)4 * 1024 * 4);
  us* GLOB  = (us*)alloc((size_t)16 * 1024 * 2);
  us* CODE1 = (us*)alloc((size_t)16 * 512 * 2);
  us* CODE  = (us*)alloc((size_t)16 * 512 * 2);
  us* DC = (us*)alloc((size_t)DR * 544 * 2);
  us* Y1 = (us*)alloc((size_t)DR * 512 * 2);
  us* Y2 = (us*)alloc((size_t)DR * 512 * 2);
  float* MID = (float*)alloc((size_t)DR * 4 * 4);
  float* REC = (float*)alloc((size_t)DR * 4 * 4);

  auto packW = [&](const void* src, us* dst, int Cout, int Cin, int CoutPad, int KPAD) {
    int tot = CoutPad * KPAD;
    pack_weight_k<<<(tot + 255) / 256, 256, 0, stream>>>((const float*)src, dst, Cout, Cin, CoutPad, KPAD);
  };
  packW(d_in[1], w1p, 64, 12, 64, 32);
  packW(d_in[3], w2p, 64, 64, 64, 64);
  packW(d_in[5], w3p, 64, 64, 64, 64);
  packW(d_in[7], g1ap, 64, 64, 64, 64);
  packW(d_in[9], g1bp, 128, 64, 128, 64);
  packW(d_in[11], g2ap, 128, 128, 128, 128);
  packW(d_in[13], g2bp, 1024, 128, 1024, 128);
  packW(d_in[15], m1p, 512, 1024, 512, 1024);
  packW(d_in[17], m2p, 512, 512, 512, 512);
  packW(d_in[19], f1ap, 512, 514, 512, 544);
  packW(d_in[21], f1bp, 512, 512, 512, 512);
  packW(d_in[23], f1cp, 3, 512, 16, 512);
  packW(d_in[25], f2ap, 512, 515, 512, 544);
  packW(d_in[27], f2bp, 512, 512, 512, 512);
  packW(d_in[29], f2cp, 3, 512, 16, 512);

  // ---- encoder ----
  knn_xyz_k<<<BN / 256, 256, 0, stream>>>(pts, idx1, N);
  local_cov_k<<<BN / 256, 256, 0, stream>>>(pts, idx1, X0, N);
  conv_wide<32, 0>(X0, w1p, b1, X1, 64, nullptr, 0, nullptr, 1, BN, 64, 64, 1, stream);
  conv_wide<64, 0>(X1, w2p, b2, X2, 64, nullptr, 0, nullptr, 1, BN, 64, 64, 1, stream);
  conv_wide<64, 0>(X2, w3p, b3, X3, 64, nullptr, 0, nullptr, 1, BN, 64, 64, 1, stream);
  sqnorm_k<<<BN / 256, 256, 0, stream>>>(X3, sqx3, 64, BN);
  knn_feat_wmma<64><<<4 * (N / 128), 256, 0, stream>>>(X3, sqx3, idx2, N);
  local_maxpool_k<<<(size_t)BN * 64 / 256, 256, 0, stream>>>(X3, idx2, P1, 64, N);
  conv_wide<64, 0>(P1, g1ap, g1ab, G1, 64, nullptr, 0, nullptr, 1, BN, 64, 64, 1, stream);
  conv_wide<64, 0>(G1, g1bp, g1bb, G, 128, nullptr, 0, nullptr, 1, BN, 128, 128, 0, stream);
  sqnorm_k<<<BN / 256, 256, 0, stream>>>(G, sqg, 128, BN);
  knn_feat_wmma<128><<<4 * (N / 128), 256, 0, stream>>>(G, sqg, idx3, N);
  local_maxpool_k<<<(size_t)BN * 128 / 256, 256, 0, stream>>>(G, idx3, P2, 128, N);
  conv_wide<128, 0>(P2, g2ap, g2ab, H1, 128, nullptr, 0, nullptr, 1, BN, 128, 128, 1, stream);
  fill_f32_k<<<(4096 + 255) / 256, 256, 0, stream>>>(globf, 4096, -3.402823466e38f);
  conv_wide<128, 2>(H1, g2bp, g2bb, nullptr, 0, nullptr, 0, globf, N, BN, 1024, 1024, 0, stream);
  glob_to_bf_k<<<16 * 1024 / 256, 256, 0, stream>>>(globf, GLOB);
  conv_small_wmma<1024><<<dim3(1, 32), 32, 0, stream>>>(GLOB, m1p, m1b, CODE1, 512, 1);
  conv_small_wmma<512><<<dim3(1, 32), 32, 0, stream>>>(CODE1, m2p, m2b, CODE, 512, 0);

  // ---- decoder: fold 1 ----
  build_fold_input_k<<<(DR + 255) / 256, 256, 0, stream>>>(CODE, nullptr, DC, 1);
  conv_wide<544, 0>(DC, f1ap, f1ab, Y1, 512, nullptr, 0, nullptr, 1, DR, 512, 512, 1, stream);
  conv_wide<512, 0>(Y1, f1bp, f1bb, Y2, 512, nullptr, 0, nullptr, 1, DR, 512, 512, 1, stream);
  conv_wide<512, 1>(Y2, f1cp, f1cb, nullptr, 0, MID, 4, nullptr, 1, DR, 3, 16, 0, stream);
  // ---- fold 2 ----
  build_fold_input_k<<<(DR + 255) / 256, 256, 0, stream>>>(CODE, MID, DC, 0);
  conv_wide<544, 0>(DC, f2ap, f2ab, Y1, 512, nullptr, 0, nullptr, 1, DR, 512, 512, 1, stream);
  conv_wide<512, 0>(Y1, f2bp, f2bb, Y2, 512, nullptr, 0, nullptr, 1, DR, 512, 512, 1, stream);
  conv_wide<512, 1>(Y2, f2cp, f2cb, nullptr, 0, REC, 4, nullptr, 1, DR, 3, 16, 0, stream);

  writeout_k<<<(4 * 2025 + 255) / 256, 256, 0, stream>>>(REC, (float*)d_out);
}
